// MyBasicNeighborAtt_15753940041865
// MI455X (gfx1250) — matrix operand order repaired and checked by reading, not compiled, not run on hardware
//
#include <hip/hip_runtime.h>
#include <hip/hip_bf16.h>

// ---------------------------------------------------------------------------
// Fused single-token neighbor-attention transformer for MI455X (gfx1250).
//
// Algebraic optimization (query has exactly 1 token):
//   scores[e,h,n] = Xn[e,n,:] . wq_eff[e,h,:] + qh[e,h,:].bk[h,:]
//     with wq_eff[e,h,d] = sum_k qh[e,h,k] * Wk[d,h,k]
//   ctx[e,h,:]    = xbar[e,h,:] @ Wv[:,h,:] + bv[h,:]
//     with xbar[e,h,:] = sum_n attn[e,h,n] * Xn[e,n,:]
// -> kh/vh (B,50,512) never materialized; ~25 GFLOP total, bound by the
//    ~213MB L2-resident neighbor gather, far below WMMA / HBM roofs.
//
// One 256-thread block (8 wave32 waves) processes BT=16 batch elements.
// Matmuls: v_wmma_f32_16x16x32_f16 (f16 operands, f32 accum).
// Fragments are loaded with b128 vector loads wherever the ISA layout is
// lane-contiguous; softmax / LayerNorm / residual stay in f32.
// ---------------------------------------------------------------------------

typedef _Float16 half_t;
typedef __attribute__((ext_vector_type(16))) _Float16 v16h;
typedef __attribute__((ext_vector_type(8)))  _Float16 v8h;
typedef __attribute__((ext_vector_type(4)))  _Float16 v4h;
typedef __attribute__((ext_vector_type(8)))  float    v8f;

#define BB    16384   // batch
#define KN    50      // neighbors
#define FD    64      // F
#define DD    64      // D
#define LL    3       // layers
#define HH    8       // heads
#define DK    64      // head dim
#define HD    512     // H*DK
#define NREF  100000
#define BT    16      // batch elements per block
#define NTHR  256     // 8 waves

// dynamic LDS (bytes): halves + floats + ints (sWqe has an 8-row zeroed pad)
#define SMEM_BYTES (((BT*64*64) + (64*96) + (BT*HD) + ((BT*8+8)*64) + (8*16*64) + \
                     (BT*8*64) + (BT*HD)) * 2 + (BT*64 + BT*64 + BT*8) * 4 + (BT*64) * 4)

// ---- WMMA fragment helpers (wave32; C/D per CDNA5 ISA 7.12.2; A/B k = 16*hf+s except loadA) ----------
// A (16x32 f16): lane L: m=L%16, hf=L/16; halves [16hf..16hf+15]
// of row m -> two b128 loads when the source row is contiguous.
__device__ __forceinline__ v16h loadA_rm(const half_t* base, int stride) {
  const int lane = threadIdx.x & 31;
  const int m = lane & 15, hf = lane >> 4;
  const half_t* p = base + m * stride + 16 * hf;
  v8h lo = *(const v8h*)p;
  v8h hi = *(const v8h*)(p + 8);
  return __builtin_shufflevector(lo, hi, 0, 1, 2, 3, 4, 5, 6, 7,
                                 8, 9, 10, 11, 12, 13, 14, 15);
}
// A from f32 row-major source (LDS qf / global X): float4 loads + pack-convert
__device__ __forceinline__ v16h loadA_rm_f32(const float* base, int stride) {
  const int lane = threadIdx.x & 31;
  const int m = lane & 15, hf = lane >> 4;
  const float* p = base + m * stride + 16 * hf;
  float4 x0 = *(const float4*)(p);
  float4 x1 = *(const float4*)(p + 4);
  float4 x2 = *(const float4*)(p + 8);
  float4 x3 = *(const float4*)(p + 12);
  v16h a;
  a[0] = (half_t)x0.x; a[1] = (half_t)x0.y; a[2]  = (half_t)x0.z; a[3]  = (half_t)x0.w;
  a[4] = (half_t)x1.x; a[5] = (half_t)x1.y; a[6]  = (half_t)x1.z; a[7]  = (half_t)x1.w;
  a[8] = (half_t)x2.x; a[9] = (half_t)x2.y; a[10] = (half_t)x2.z; a[11] = (half_t)x2.w;
  a[12] = (half_t)x3.x; a[13] = (half_t)x3.y; a[14] = (half_t)x3.z; a[15] = (half_t)x3.w;
  return a;
}
// generic scalar A (fallback)
template <typename F>
__device__ __forceinline__ v16h loadA(F f) {
  const int lane = threadIdx.x & 31;
  const int m = lane & 15, hf = lane >> 4;
  v16h a;
#pragma unroll
  for (int v = 0; v < 8; ++v) {
    const int kb = (v < 4) ? (8 * hf + 2 * v) : (16 + 8 * hf + 2 * (v - 4));
    a[2 * v]     = f(m, kb);
    a[2 * v + 1] = f(m, kb + 1);
  }
  return a;
}
// B (32x16 f16): lane L: n=L%16, hf=L/16; VGPR v -> k = 16*hf + 2v{,+1}
// When B is stored K-contiguous per column (element (k,n) at base+n*stride+k),
// a lane needs 16 consecutive halves starting at 16*hf -> two b128 loads.
__device__ __forceinline__ v16h loadBt(const half_t* base, int stride) {
  const int lane = threadIdx.x & 31;
  const int n = lane & 15, hf = lane >> 4;
  const half_t* p = base + n * stride + 16 * hf;
  v8h lo = *(const v8h*)p;
  v8h hi = *(const v8h*)(p + 8);
  return __builtin_shufflevector(lo, hi, 0, 1, 2, 3, 4, 5, 6, 7,
                                 8, 9, 10, 11, 12, 13, 14, 15);
}
// same but f32 source (transposed-read global weights, e.g. Wk[d,h,:])
__device__ __forceinline__ v16h loadBt_f32(const float* base, long stride) {
  const int lane = threadIdx.x & 31;
  const int n = lane & 15, hf = lane >> 4;
  const float* p = base + (long)n * stride + 16 * hf;
  float4 x0 = *(const float4*)(p);
  float4 x1 = *(const float4*)(p + 4);
  float4 x2 = *(const float4*)(p + 8);
  float4 x3 = *(const float4*)(p + 12);
  v16h b;
  b[0] = (half_t)x0.x; b[1] = (half_t)x0.y; b[2]  = (half_t)x0.z; b[3]  = (half_t)x0.w;
  b[4] = (half_t)x1.x; b[5] = (half_t)x1.y; b[6]  = (half_t)x1.z; b[7]  = (half_t)x1.w;
  b[8] = (half_t)x2.x; b[9] = (half_t)x2.y; b[10] = (half_t)x2.z; b[11] = (half_t)x2.w;
  b[12] = (half_t)x3.x; b[13] = (half_t)x3.y; b[14] = (half_t)x3.z; b[15] = (half_t)x3.w;
  return b;
}
// generic scalar B (row-major / guarded sources)
template <typename F>
__device__ __forceinline__ v16h loadB(F f) {
  const int lane = threadIdx.x & 31;
  const int n = lane & 15, hf = lane >> 4;
  v16h b;
#pragma unroll
  for (int v = 0; v < 8; ++v) {
    const int kb = 16 * hf + 2 * v;
    b[2 * v]     = f(kb, n);
    b[2 * v + 1] = f(kb + 1, n);
  }
  return b;
}
// C/D (16x16 f32): VGPR r: m = r + 8*(L/16), n = L%16
template <typename F>
__device__ __forceinline__ void storeC(const v8f& c, F f) {
  const int lane = threadIdx.x & 31;
  const int n = lane & 15, hf = lane >> 4;
#pragma unroll
  for (int r = 0; r < 8; ++r) f(r + 8 * hf, n, c[r]);
}

__device__ __forceinline__ v8f wmma(v16h a, v16h b, v8f c) {
  return __builtin_amdgcn_wmma_f32_16x16x32_f16(false, a, false, b, (short)0, c,
                                                false, false);
}

__global__ __launch_bounds__(NTHR)
void natt_kernel(const float* __restrict__ X, const int* __restrict__ S,
                 const float* __restrict__ X_ref, const float* __restrict__ y_ref,
                 const float* __restrict__ W_emb, const float* __restrict__ b_emb,
                 const float* __restrict__ W_nemb, const float* __restrict__ b_nemb,
                 const float* __restrict__ Wq, const float* __restrict__ bq,
                 const float* __restrict__ Wk, const float* __restrict__ bk,
                 const float* __restrict__ Wv, const float* __restrict__ bv,
                 const float* __restrict__ Wo, const float* __restrict__ bo,
                 const float* __restrict__ ln_g, const float* __restrict__ ln_b,
                 const float* __restrict__ W1, const float* __restrict__ b1,
                 const float* __restrict__ W2, const float* __restrict__ b2,
                 const float* __restrict__ W3, const float* __restrict__ b3,
                 float* __restrict__ out)
{
  extern __shared__ char smem[];
  half_t* sXn   = (half_t*)smem;            // [BT][64][64] padded neighbor acts
  half_t* sRaw  = sXn   + BT * 64 * 64;     // [64][96] gather staging (1 elem)
  half_t* sQh   = sRaw  + 64 * 96;          // [BT][512] qh
  half_t* sWqe  = sQh   + BT * HD;          // [BT][8][64] wq_eff (+8 pad rows)
  half_t* sAttn = sWqe  + (BT * 8 + 8) * 64;// [8 waves][16][64] attn (A-op)
  half_t* sXbar = sAttn + 8 * 16 * 64;      // [BT][8][64]
  half_t* sCtx  = sXbar + BT * 8 * 64;      // [BT][512]
  float*  qf    = (float*)(sCtx + BT * HD); // [BT][64] residual stream (f32)
  float*  sTmp  = qf   + BT * 64;           // [BT][64] attn_out scratch
  float*  qbkS  = sTmp + BT * 64;           // [BT][8]  qh.bk scalars
  int*    sS    = (int*)(qbkS + BT * 8);    // [BT][64] neighbor ids (50 used)

  const int tid  = threadIdx.x;
  const int lane = tid & 31;
  const int wv   = tid >> 5;                // wave id 0..7
  const int b0   = blockIdx.x * BT;

  // ---- stage neighbor indices; zero pads once (never overwritten) ----
  for (int i = tid; i < BT * KN; i += NTHR) {
    const int e = i / KN, n = i % KN;
    sS[e * 64 + n] = S[(long)(b0 + e) * KN + n];
  }
  for (int i = tid; i < 64 * 96; i += NTHR) sRaw[i] = (half_t)0.f;
  for (int i = tid; i < 8 * 64; i += NTHR) sWqe[BT * 8 * 64 + i] = (half_t)0.f;
  __syncthreads();

  // ---- Phase A: gather + neighbor embed  Xn = relu([Xref|y] @ W_nemb) -----
  for (int e = 0; e < BT; ++e) {
    // 50x64 X_ref gather at float4 granularity, f16 b64 stores
    for (int i = tid; i < KN * (FD / 4); i += NTHR) {
      const int n = i >> 4, k4 = (i & 15) * 4;
      const int s = sS[e * 64 + n];
      const int c = s < 0 ? 0 : (s >= NREF ? NREF - 1 : s);
      const float4 v = *(const float4*)(X_ref + (long)c * FD + k4);
      v4h h4;
      h4[0] = (half_t)v.x; h4[1] = (half_t)v.y;
      h4[2] = (half_t)v.z; h4[3] = (half_t)v.w;
      *(v4h*)(sRaw + n * 96 + k4) = h4;
    }
    for (int n = tid; n < KN; n += NTHR) {   // y column (col 64)
      const int s = sS[e * 64 + n];
      const int c = s < 0 ? 0 : (s >= NREF ? NREF - 1 : s);
      sRaw[n * 96 + FD] = (half_t)y_ref[c];
    }
    __syncthreads();
#pragma unroll
    for (int t0 = 0; t0 < 2; ++t0) {         // 4 Mtiles x 4 Ntiles / 8 waves
      const int t = wv + 8 * t0;
      const int mt = t >> 2, nt = t & 3;
      v8f acc = {};
#pragma unroll
      for (int kc = 0; kc < 3; ++kc) {       // K padded 65 -> 96
        v16h a = loadA_rm(sRaw + (mt * 16) * 96 + kc * 32, 96);
        v16h b = loadB([&](int k, int n) {
          const int kk = kc * 32 + k;
          return (half_t)(kk < FD + 1 ? W_nemb[kk * DD + nt * 16 + n] : 0.f); });
        acc = wmma(a, b, acc);
      }
      storeC(acc, [&](int m, int n, float c) {
        const int row = mt * 16 + m, col = nt * 16 + n;
        float v = c + b_nemb[col];
        v = v > 0.f ? v : 0.f;
        sXn[(e * 64 + row) * 64 + col] = (half_t)(row < KN ? v : 0.f);
      });
    }
    __syncthreads();
  }

  // ---- Phase B: q = relu(X @ W_emb + b_emb) -------------------------------
  if (wv < 4) {
    const int nt = wv;
    v8f acc = {};
#pragma unroll
    for (int kc = 0; kc < 2; ++kc) {
      v16h a = loadA_rm_f32(X + (long)b0 * FD + kc * 32, FD);
      v16h b = loadB([&](int k, int n) {
        return (half_t)W_emb[(kc * 32 + k) * DD + nt * 16 + n]; });
      acc = wmma(a, b, acc);
    }
    storeC(acc, [&](int m, int n, float c) {
      const float v = c + b_emb[nt * 16 + n];
      qf[m * DD + nt * 16 + n] = v > 0.f ? v : 0.f;
    });
  }
  __syncthreads();

  // ---- Layers -------------------------------------------------------------
  for (int li = 0; li < LL; ++li) {
    const float* Wq_i = Wq + (long)li * DD * HD;
    const float* bq_i = bq + li * HD;
    const float* Wk_i = Wk + (long)li * DD * HD;
    const float* bk_i = bk + li * HD;
    const float* Wv_i = Wv + (long)li * DD * HD;
    const float* bv_i = bv + li * HD;
    const float* Wo_i = Wo + (long)li * HD * DD;
    const float* bo_i = bo + li * DD;
    const float* g_i  = ln_g + li * DD;
    const float* be_i = ln_b + li * DD;

    // warm this layer's weights into cache (global_prefetch_b8)
    for (int off = tid * 64; off < DD * HD; off += NTHR * 64) {
      __builtin_prefetch(Wq_i + off, 0, 0);
      __builtin_prefetch(Wk_i + off, 0, 0);
      __builtin_prefetch(Wv_i + off, 0, 0);
      __builtin_prefetch(Wo_i + off, 0, 0);
    }

    // Phase C: qh = q @ Wq + bq   (16 x 512)
#pragma unroll
    for (int j = 0; j < 4; ++j) {
      const int nt = wv * 4 + j;
      v8f acc = {};
#pragma unroll
      for (int kc = 0; kc < 2; ++kc) {
        v16h a = loadA_rm_f32(qf + kc * 32, DD);
        v16h b = loadB([&](int k, int n) {
          return (half_t)Wq_i[(kc * 32 + k) * HD + nt * 16 + n]; });
        acc = wmma(a, b, acc);
      }
      storeC(acc, [&](int m, int n, float c) {
        const int col = nt * 16 + n;
        sQh[m * HD + col] = (half_t)(c + bq_i[col]);
      });
    }
    __syncthreads();

    // Phase D: wq_eff[e,h,d] = sum_k qh[e,h,k]*Wk[d,h,k]  (one head per wave)
    {
      const int h = wv;
#pragma unroll
      for (int nt = 0; nt < 4; ++nt) {
        v8f acc = {};
#pragma unroll
        for (int kc = 0; kc < 2; ++kc) {
          v16h a = loadA_rm(sQh + h * DK + kc * 32, HD);
          // B[k][d] = Wk[d,h,k]: 16 consecutive k per lane -> float4 loads
          v16h b = loadBt_f32(Wk_i + (long)(nt * 16) * HD + h * DK + kc * 32, HD);
          acc = wmma(a, b, acc);
        }
        storeC(acc, [&](int m, int n, float c) {
          sWqe[(m * 8 + h) * DK + nt * 16 + n] = (half_t)c;
        });
      }
    }
    if (tid < BT * HH) {  // qbk[e,h] = qh[e,h,:].bk[h,:]
      const int e = tid >> 3, h = tid & 7;
      float s = 0.f;
      for (int k = 0; k < DK; ++k)
        s += (float)sQh[e * HD + h * DK + k] * bk_i[h * DK + k];
      qbkS[e * HH + h] = s;
    }
    __syncthreads();

    // Phase E: scores -> softmax (registers) -> attn -> xbar  (2 elems/wave)
    for (int p = 0; p < 2; ++p) {
      const int e = wv + 8 * p;
      v8f cs[4];
#pragma unroll
      for (int mt = 0; mt < 4; ++mt) {      // scores: (64n x 64d)@(64d x 16h)
        v8f acc = {};
#pragma unroll
        for (int kc = 0; kc < 2; ++kc) {
          v16h a = loadA_rm(sXn + (e * 64 + mt * 16) * 64 + kc * 32, 64);
          // B[k][h] = wq_eff[e,h,k]: K-contiguous -> b128 loads
          // (columns h>=8 read pad/neighbor rows; discarded below)
          v16h b = loadBt(sWqe + e * 8 * 64 + kc * 32, 64);
          acc = wmma(a, b, acc);
        }
        cs[mt] = acc;
      }
      // softmax over neighbors per head; head h = lane%16, halves via xor-16
      const int h = lane & 15, hf = lane >> 4;
      const float qb = (h < HH) ? qbkS[e * HH + h] : 0.f;
      const float scale = 0.125f;           // 1/sqrt(64)
      float mx = -3.0e38f;
#pragma unroll
      for (int mt = 0; mt < 4; ++mt)
#pragma unroll
        for (int r = 0; r < 8; ++r) {
          const int m = mt * 16 + 8 * hf + r;
          const bool valid = (m < KN) && (sS[e * 64 + m] >= 0);
          const float sc = valid ? cs[mt][r] * scale + qb : -1.0e9f;
          cs[mt][r] = sc;
          mx = sc > mx ? sc : mx;
        }
      mx = fmaxf(mx, __shfl_xor(mx, 16, 32));
      float sum = 0.f;
#pragma unroll
      for (int mt = 0; mt < 4; ++mt)
#pragma unroll
        for (int r = 0; r < 8; ++r) {
          const float ex = __expf(cs[mt][r] - mx);
          cs[mt][r] = ex;
          sum += ex;
        }
      sum += __shfl_xor(sum, 16, 32);
      const float inv = 1.f / sum;
#pragma unroll
      for (int mt = 0; mt < 4; ++mt)
#pragma unroll
        for (int r = 0; r < 8; ++r) {
          const int m = mt * 16 + 8 * hf + r;
          sAttn[(wv * 16 + h) * 64 + m] =
              (half_t)((h < HH) ? cs[mt][r] * inv : 0.f);
        }
      __syncthreads();
      // xbar[e,h,:] = attn @ Xn[e]
#pragma unroll
      for (int nt = 0; nt < 4; ++nt) {
        v8f acc = {};
#pragma unroll
        for (int kc = 0; kc < 2; ++kc) {
          v16h a = loadA_rm(sAttn + (wv * 16) * 64 + kc * 32, 64);
          v16h b = loadB([&](int k, int n) {
            return sXn[(e * 64 + kc * 32 + k) * 64 + nt * 16 + n]; });
          acc = wmma(a, b, acc);
        }
        storeC(acc, [&](int m, int n, float c) {
          if (m < HH) sXbar[(e * 8 + m) * DK + nt * 16 + n] = (half_t)c;
        });
      }
      __syncthreads();
    }

    // Phase F: ctx[e,h,:] = xbar[e,h,:] @ Wv[:,h,:] + bv   (one head per wave)
    {
      const int h = wv;
#pragma unroll
      for (int nt = 0; nt < 4; ++nt) {
        v8f acc = {};
#pragma unroll
        for (int kc = 0; kc < 2; ++kc) {
          v16h a = loadA_rm(sXbar + h * DK + kc * 32, 8 * DK);
          v16h b = loadB([&](int k, int n) {
            return (half_t)Wv_i[(long)(kc * 32 + k) * HD + h * DK + nt * 16 + n]; });
          acc = wmma(a, b, acc);
        }
        storeC(acc, [&](int m, int n, float c) {
          const int col = h * DK + nt * 16 + n;
          sCtx[m * HD + col] = (half_t)(c + bv_i[col]);
        });
      }
    }
    __syncthreads();

    // Phase G: attn_out = ctx @ Wo + bo ; residual + LayerNorm (f32)
    if (wv < 4) {
      const int nt = wv;
      v8f acc = {};
#pragma unroll 4
      for (int kc = 0; kc < 16; ++kc) {     // K = 512
        v16h a = loadA_rm(sCtx + kc * 32, HD);
        v16h b = loadB([&](int k, int n) {
          return (half_t)Wo_i[(long)(kc * 32 + k) * DD + nt * 16 + n]; });
        acc = wmma(a, b, acc);
      }
      storeC(acc, [&](int m, int n, float c) {
        const int col = nt * 16 + n;
        sTmp[m * DD + col] = c + bo_i[col];
      });
    }
    __syncthreads();
    if (tid < BT) {
      const int e = tid;
      float mu = 0.f;
      for (int d = 0; d < DD; ++d) mu += qf[e * DD + d] + sTmp[e * DD + d];
      mu *= (1.f / DD);
      float var = 0.f;
      for (int d = 0; d < DD; ++d) {
        const float x = qf[e * DD + d] + sTmp[e * DD + d] - mu;
        var += x * x;
      }
      var *= (1.f / DD);
      const float inv = rsqrtf(var + 1e-3f);
      for (int d = 0; d < DD; ++d) {
        const float x = qf[e * DD + d] + sTmp[e * DD + d];
        qf[e * DD + d] = (x - mu) * inv * g_i[d] + be_i[d];
      }
    }
    __syncthreads();
  } // layers

  // ---- Phase H: MLP head --------------------------------------------------
  half_t* sH1 = sQh;            // reuse
  half_t* sH2 = sQh + BT * DD;
  if (wv < 4) {
    const int nt = wv;
    v8f acc = {};
#pragma unroll
    for (int kc = 0; kc < 2; ++kc) {
      v16h a = loadA_rm_f32(qf + kc * 32, DD);
      v16h b = loadB([&](int k, int n) {
        return (half_t)W1[(kc * 32 + k) * DD + nt * 16 + n]; });
      acc = wmma(a, b, acc);
    }
    storeC(acc, [&](int m, int n, float c) {
      const int col = nt * 16 + n;
      const float v = c + b1[col];
      sH1[m * DD + col] = (half_t)(v > 0.f ? v : 0.f);
    });
  }
  __syncthreads();
  if (wv < 4) {
    const int nt = wv;
    v8f acc = {};
#pragma unroll
    for (int kc = 0; kc < 2; ++kc) {
      v16h a = loadA_rm(sH1 + kc * 32, DD);
      v16h b = loadB([&](int k, int n) {
        return (half_t)W2[(kc * 32 + k) * DD + nt * 16 + n]; });
      acc = wmma(a, b, acc);
    }
    storeC(acc, [&](int m, int n, float c) {
      const int col = nt * 16 + n;
      const float v = c + b2[col];
      sH2[m * DD + col] = (half_t)(v > 0.f ? v : 0.f);
    });
  }
  __syncthreads();
  if (tid < BT) {
    float s = 0.f;
    for (int d = 0; d < DD; ++d) s += (float)sH2[tid * DD + d] * W3[d];
    out[b0 + tid] = s + b3[0];
  }
}

extern "C" void kernel_launch(void* const* d_in, const int* in_sizes, int n_in,
                              void* d_out, int out_size, void* d_ws, size_t ws_size,
                              hipStream_t stream) {
  (void)in_sizes; (void)n_in; (void)out_size; (void)d_ws; (void)ws_size;
  const float* X      = (const float*)d_in[0];
  const int*   S      = (const int*)  d_in[1];
  const float* X_ref  = (const float*)d_in[2];
  const float* y_ref  = (const float*)d_in[3];
  const float* W_emb  = (const float*)d_in[4];
  const float* b_emb  = (const float*)d_in[5];
  const float* W_nemb = (const float*)d_in[6];
  const float* b_nemb = (const float*)d_in[7];
  const float* Wq     = (const float*)d_in[8];
  const float* bq     = (const float*)d_in[9];
  const float* Wk     = (const float*)d_in[10];
  const float* bk     = (const float*)d_in[11];
  const float* Wv     = (const float*)d_in[12];
  const float* bv     = (const float*)d_in[13];
  const float* Wo     = (const float*)d_in[14];
  const float* bo     = (const float*)d_in[15];
  const float* ln_g   = (const float*)d_in[16];
  const float* ln_b   = (const float*)d_in[17];
  const float* W1     = (const float*)d_in[18];
  const float* b1     = (const float*)d_in[19];
  const float* W2     = (const float*)d_in[20];
  const float* b2     = (const float*)d_in[21];
  const float* W3     = (const float*)d_in[22];
  const float* b3     = (const float*)d_in[23];
  float* out = (float*)d_out;

  // Allow >64KB dynamic LDS (gfx1250 supports up to 320KB per workgroup).
  (void)hipFuncSetAttribute(reinterpret_cast<const void*>(natt_kernel),
                            hipFuncAttributeMaxDynamicSharedMemorySize,
                            (int)SMEM_BYTES);

  natt_kernel<<<BB / BT, NTHR, SMEM_BYTES, stream>>>(
      X, S, X_ref, y_ref, W_emb, b_emb, W_nemb, b_nemb,
      Wq, bq, Wk, bk, Wv, bv, Wo, bo, ln_g, ln_b,
      W1, b1, W2, b2, W3, b3, out);
}
